// AttentionLayer_42649025249335
// MI455X (gfx1250) — compile-verified, matrix operand order checked by reading, not run on hardware
//
#include <hip/hip_runtime.h>

#define T_DIM 2048
#define B_DIM 2
#define NIN   1024
#define H_DIM 16
#define D_HEAD 64

typedef _Float16 F16;
typedef F16   v16h __attribute__((ext_vector_type(16)));
typedef F16   v8h  __attribute__((ext_vector_type(8)));
typedef float v8f  __attribute__((ext_vector_type(8)));

// Load 16 f16 in the WMMA 16-bit A/B per-lane pattern: 8 contiguous at p,
// 8 contiguous at p+16 (two global_load_b128).
__device__ __forceinline__ v16h load_ab(const F16* p) {
  v8h lo = *(const v8h*)(p);
  v8h hi = *(const v8h*)(p + 16);
  return __builtin_shufflevector(lo, hi, 0,1,2,3,4,5,6,7,8,9,10,11,12,13,14,15);
}

__device__ __forceinline__ v8f wmma16(v16h a, v16h b, v8f c) {
  return __builtin_amdgcn_wmma_f32_16x16x32_f16(false, a, false, b, (short)0, c,
                                                false, false);
}

__global__ void cvt_f32_to_f16(const float* __restrict__ in, F16* __restrict__ out, int n) {
  int stride = gridDim.x * blockDim.x;
  for (int i = blockIdx.x * blockDim.x + threadIdx.x; i < n; i += stride)
    out[i] = (F16)in[i];
}

// Y = X(4096x1024,f16) @ W^T(1024x1024,f16) + bias, scaled, stored f16.
// Each wave: one 16-row x 64-col strip (4 accumulator tiles), K looped by 32,
// software-pipelined (loads lead WMMAs by one iteration).
// Block-mates share the same W strip (identical B addresses -> WGP$ reuse).
// transposeV=1 stores per-head transposed [b][h][d][t] (for the P@V B-operand).
__global__ void proj_gemm(const F16* __restrict__ xh, const F16* __restrict__ wh,
                          const float* __restrict__ bias, F16* __restrict__ out,
                          float scale, int transposeV) {
  const int lane = threadIdx.x & 31;
  const int wave = (blockIdx.x * blockDim.x + threadIdx.x) >> 5;  // 0..4095
  const int l16  = lane & 15;
  const int hoff = (lane & 16) ? 8 : 0;     // K-pattern half offset (also row half)
  // block = 8 waves: same W strip, 8 consecutive m-tiles
  const int strip = (wave >> 3) & 15;       // shared within block
  const int m0 = (((wave >> 7) << 3) + (wave & 7)) << 4;  // 256 m-tiles
  const int n0 = strip << 6;                // 16 strips of 64 cols (one head)

  v8f acc0 = {}; v8f acc1 = {}; v8f acc2 = {}; v8f acc3 = {};
  const F16* xrow = xh + (size_t)(m0 + l16) * NIN + hoff;
  const F16* w0 = wh + (size_t)(n0 +  0 + l16) * NIN + hoff;
  const F16* w1 = wh + (size_t)(n0 + 16 + l16) * NIN + hoff;
  const F16* w2 = wh + (size_t)(n0 + 32 + l16) * NIN + hoff;
  const F16* w3 = wh + (size_t)(n0 + 48 + l16) * NIN + hoff;

  // pipeline prologue
  v16h a_c  = load_ab(xrow);
  v16h b0_c = load_ab(w0);
  v16h b1_c = load_ab(w1);
  v16h b2_c = load_ab(w2);
  v16h b3_c = load_ab(w3);

  for (int j = 0; j < NIN; j += 32) {
    const int jn = (j + 32 < NIN) ? (j + 32) : j;   // clamp: last iter reloads
    v16h a_n  = load_ab(xrow + jn);
    v16h b0_n = load_ab(w0 + jn);
    v16h b1_n = load_ab(w1 + jn);
    v16h b2_n = load_ab(w2 + jn);
    v16h b3_n = load_ab(w3 + jn);
    acc0 = wmma16(a_c, b0_c, acc0);
    acc1 = wmma16(a_c, b1_c, acc1);
    acc2 = wmma16(a_c, b2_c, acc2);
    acc3 = wmma16(a_c, b3_c, acc3);
    a_c = a_n; b0_c = b0_n; b1_c = b1_n; b2_c = b2_n; b3_c = b3_n;
  }
#pragma unroll
  for (int nt = 0; nt < 4; ++nt) {
    v8f acc = nt == 0 ? acc0 : nt == 1 ? acc1 : nt == 2 ? acc2 : acc3;
    const int n  = n0 + nt * 16 + l16;   // global output column (lane = N)
    const int hh = n >> 6, d = n & 63;
    const float bvv = bias[n];
#pragma unroll
    for (int v = 0; v < 8; ++v) {        // VGPR index = row within tile
      int m = m0 + v + hoff;             // hoff == 8*(lane/16)
      int t = m >> 1, b = m & 1;         // x rows are (t,b) interleaved, B=2
      float y = (acc[v] + bvv) * scale;
      if (transposeV)
        out[((size_t)(b * H_DIM + hh) * D_HEAD + d) * T_DIM + t] = (F16)y;
      else
        out[((size_t)(b * H_DIM + hh) * T_DIM + t) * D_HEAD + d] = (F16)y;
    }
  }
}

// Flash attention, softmax over k (query axis). keys carry log2(e)/sqrt(NIN).
// Each wave: 16 output rows i for one (b,h); streams k in blocks of 32.
// S^T tiles are computed so P lands directly in WMMA A-layout (no transpose).
// Q A-operands double-buffered; V loads hoisted above S-WMMAs + softmax.
__global__ void attn_fa(const F16* __restrict__ keys, const F16* __restrict__ queries,
                        const F16* __restrict__ valT, const unsigned char* __restrict__ mask,
                        float* __restrict__ out) {
  const int lane = threadIdx.x & 31;
  const int wave = (blockIdx.x * blockDim.x + threadIdx.x) >> 5;
  const int l16  = lane & 15;
  const int hoff = (lane & 16) ? 8 : 0;

  const int itile = wave & 127;
  const int bh    = wave >> 7;            // 0..31 (block-mates share bh -> Q/V reuse)
  const int b = bh >> 4, hh = bh & 15;
  const int i0 = itile << 4;

  const F16* kp = keys    + (size_t)bh * T_DIM * D_HEAD;   // [t][d]
  const F16* qp = queries + (size_t)bh * T_DIM * D_HEAD;   // [t][d]
  const F16* vp = valT    + (size_t)bh * D_HEAD * T_DIM;   // [d][t]
  const unsigned char* mp = mask + b * T_DIM;

  // Loop-invariant B-operand: keys rows i0..i0+15 (lane = column i).
  v16h bk0 = load_ab(kp + (size_t)(i0 + l16) * D_HEAD +  0 + hoff);
  v16h bk1 = load_ab(kp + (size_t)(i0 + l16) * D_HEAD + 32 + hoff);

  v8f o0 = {}, o1 = {}, o2 = {}, o3 = {};
  float lsum = 0.f;

  // pipeline prologue: A-operand (queries rows) for kb = 0
  v16h aq0 = load_ab(qp + (size_t)(l16) * D_HEAD + hoff);
  v16h aq1 = load_ab(qp + (size_t)(l16) * D_HEAD + 32 + hoff);
  v16h aq2 = load_ab(qp + (size_t)(16 + l16) * D_HEAD + hoff);
  v16h aq3 = load_ab(qp + (size_t)(16 + l16) * D_HEAD + 32 + hoff);

  for (int kb = 0; kb < T_DIM; kb += 32) {
    // V-operand loads for current kb; complete under S-WMMAs + softmax
    v16h vv0 = load_ab(vp + (size_t)( 0 + l16) * T_DIM + kb + hoff);
    v16h vv1 = load_ab(vp + (size_t)(16 + l16) * T_DIM + kb + hoff);
    v16h vv2 = load_ab(vp + (size_t)(32 + l16) * T_DIM + kb + hoff);
    v16h vv3 = load_ab(vp + (size_t)(48 + l16) * T_DIM + kb + hoff);

    v8f s0 = {}; v8f s1 = {};
    s0 = wmma16(aq0, bk0, s0);     // S^T[k, i], k = kb..kb+15
    s0 = wmma16(aq1, bk1, s0);
    s1 = wmma16(aq2, bk0, s1);     // k = kb+16..kb+31
    s1 = wmma16(aq3, bk1, s1);

    // issue next-iteration A loads now (consumed next iter)
    const int kn = (kb + 32 < T_DIM) ? (kb + 32) : kb;
    v16h na0 = load_ab(qp + (size_t)(kn + l16) * D_HEAD + hoff);
    v16h na1 = load_ab(qp + (size_t)(kn + l16) * D_HEAD + 32 + hoff);
    v16h na2 = load_ab(qp + (size_t)(kn + 16 + l16) * D_HEAD + hoff);
    v16h na3 = load_ab(qp + (size_t)(kn + 16 + l16) * D_HEAD + 32 + hoff);

    // stream prefetch two k-blocks ahead (global_prefetch_b8)
    const int kp2 = (kb + 64 < T_DIM) ? (kb + 64) : kb;
    __builtin_prefetch(qp + (size_t)(kp2 + lane) * D_HEAD, 0, 1);
    __builtin_prefetch(vp + (size_t)(2 * lane) * T_DIM + kp2, 0, 1);

    // mask bytes for this lane's 8 k-rows per tile (8B aligned)
    unsigned long long mb0 = *(const unsigned long long*)(mp + kb + hoff);
    unsigned long long mb1 = *(const unsigned long long*)(mp + kb + 16 + hoff);

    v16h ap;                 // P in A-layout: rows i = lane%16, cols k pattern
    float bsum = 0.f;
#pragma unroll
    for (int j = 0; j < 8; ++j) {
      float sv = s0[j] + (((mb0 >> (8 * j)) & 0xffull) ? 0.f : -1e30f);
      float p = exp2f(sv);   // log2(e) folded into keys; |S|~O(1), no max needed
      bsum += p;
      ap[j] = (F16)p;
    }
#pragma unroll
    for (int j = 0; j < 8; ++j) {
      float sv = s1[j] + (((mb1 >> (8 * j)) & 0xffull) ? 0.f : -1e30f);
      float p = exp2f(sv);
      bsum += p;
      ap[8 + j] = (F16)p;
    }
    bsum += __shfl_xor(bsum, 16, 32);   // merge the two k-halves per row i
    lsum += bsum;

    // O[i, d] += P[i, k] * V[k, d]  (V^T layout -> contiguous b128 loads)
    o0 = wmma16(ap, vv0, o0);
    o1 = wmma16(ap, vv1, o1);
    o2 = wmma16(ap, vv2, o2);
    o3 = wmma16(ap, vv3, o3);

    aq0 = na0; aq1 = na1; aq2 = na2; aq3 = na3;
  }

  float rinv = 1.0f / lsum;             // stat lives at lanes with l16 == row
#pragma unroll
  for (int v = 0; v < 8; ++v) {
    float rv = __shfl(rinv, v + hoff, 32);     // row r = v + 8*(lane/16)
    int t = i0 + v + hoff;
    size_t base = ((size_t)t * B_DIM + b) * (H_DIM * D_HEAD) + hh * D_HEAD + l16;
    out[base +  0] = o0[v] * rv;
    out[base + 16] = o1[v] * rv;
    out[base + 32] = o2[v] * rv;
    out[base + 48] = o3[v] * rv;
  }
}

extern "C" void kernel_launch(void* const* d_in, const int* in_sizes, int n_in,
                              void* d_out, int out_size, void* d_ws, size_t ws_size,
                              hipStream_t stream) {
  (void)in_sizes; (void)n_in; (void)out_size; (void)ws_size;
  const float* x  = (const float*)d_in[0];
  const unsigned char* mask = (const unsigned char*)d_in[1];
  const float* Wk = (const float*)d_in[2];
  const float* bk = (const float*)d_in[3];
  const float* Wq = (const float*)d_in[4];
  const float* bq = (const float*)d_in[5];
  const float* Wv = (const float*)d_in[6];
  const float* bv = (const float*)d_in[7];
  float* out = (float*)d_out;

  char* ws = (char*)d_ws;                       // 38 MB carved, 16B aligned
  F16* xh  = (F16*)(ws);                        // 8 MB: x as f16
  F16* wkh = (F16*)(ws + ( 8u << 20));          // 2 MB each
  F16* wqh = (F16*)(ws + (10u << 20));
  F16* wvh = (F16*)(ws + (12u << 20));
  F16* kh  = (F16*)(ws + (14u << 20));          // keys [b][h][t][d], pre-scaled
  F16* qh  = (F16*)(ws + (22u << 20));          // queries [b][h][t][d]
  F16* vth = (F16*)(ws + (30u << 20));          // values^T [b][h][d][t]

  cvt_f32_to_f16<<<2048, 256, 0, stream>>>(x,  xh,  T_DIM * B_DIM * NIN);
  cvt_f32_to_f16<<<1024, 256, 0, stream>>>(Wk, wkh, NIN * NIN);
  cvt_f32_to_f16<<<1024, 256, 0, stream>>>(Wq, wqh, NIN * NIN);
  cvt_f32_to_f16<<<1024, 256, 0, stream>>>(Wv, wvh, NIN * NIN);

  const float kscale = 1.4426950408889634f / 32.0f;  // log2(e)/sqrt(NIN)
  proj_gemm<<<512, 256, 0, stream>>>(xh, wkh, bk, kh,  kscale, 0);
  proj_gemm<<<512, 256, 0, stream>>>(xh, wqh, bq, qh,  1.0f,   0);
  proj_gemm<<<512, 256, 0, stream>>>(xh, wvh, bv, vth, 1.0f,   1);

  attn_fa<<<512, 256, 0, stream>>>(kh, qh, vth, mask, out);
}
